// SymmetricContraction_56349970923982
// MI455X (gfx1250) — compile-verified, matrix operand order checked by reading, not run on hardware
//
#include <hip/hip_runtime.h>
#include <hip/hip_bf16.h>

// MACE symmetric contraction for MI455X (gfx1250, wave32, WMMA).
//
// Strategy: fold U*W into per-channel bf16 matrices G_c (K=192 x N=1088,
// pre-swizzled into the WMMA B-fragment lane layout), build
// X[b,(e,i)] = attrs[b,e]*A[b,c,i] in LDS in the A-fragment layout, and run
// the correlation-3 contraction (+fused level-2/level-1 init terms) as a
// dense bf16 WMMA GEMM with f32 accumulation. The j2/j1 contractions with A
// are tiny per-tile VALU epilogues using wave32 shfl_xor reductions.
//
// Workspace: G_swz bf16, C*68*6*32*16 elements = 53,477,376 bytes.

typedef __attribute__((ext_vector_type(16))) __bf16 v16bf;
typedef __attribute__((ext_vector_type(8)))  float  v8f;

#define NB 1024
#define NC 128
#define NI 16
#define NE 10
#define NT 68   // n-tiles of 16: 16 scalar jj + 48 vector jj + 1 P1s + 3 P1v
#define NKS 6   // k-steps of 32: K = 192 (160 (e,i) rows + 10 attr rows + pad)

// K index inside a 32-wide k-step for (lane, half h) per ISA 16-bit fragment
// layout: VGPR j holds K pair (2j,2j+1) for j<4 (+16 for j>=4), lanes>=16 add 8.
__device__ __forceinline__ int klocal(int lane, int h) {
  return h + (h & 8) + ((lane & 16) >> 1);
}

// ---------------------------------------------------------------------------
// Kernel 1: densify U*W -> G, pre-swizzled into WMMA B-fragment layout.
// grid = NC*NT*NKS blocks x 512 threads (32 lanes x 16 halves each).
// ---------------------------------------------------------------------------
__global__ void symcon_g_precompute(
    const float* __restrict__ U0_1, const float* __restrict__ U0_2,
    const float* __restrict__ U0_3, const float* __restrict__ U1_1,
    const float* __restrict__ U1_2, const float* __restrict__ U1_3,
    const float* __restrict__ W0_1, const float* __restrict__ W0_2,
    const float* __restrict__ W0_3, const float* __restrict__ W1_1,
    const float* __restrict__ W1_2, const float* __restrict__ W1_3,
    __bf16* __restrict__ G)
{
  int blk = blockIdx.x;
  int ks = blk % NKS;
  int tmp = blk / NKS;
  int t = tmp % NT;
  int c = tmp / NT;
  int h = threadIdx.x & 15;
  int lane = threadIdx.x >> 4;

  int r = ks * 32 + klocal(lane, h);   // row of G matrix (K dim), 0..191
  int n = t * 16 + (lane & 15);        // column of G matrix, 0..1087

  float v = 0.f;
  if (r < 160) {
    // rows (e,i): level-3 term  sum_k U3[jj,i,k] * W3[e,k,c]
    if (n < 1024) {
      int e = r >> 4, i = r & 15;
      if (n < 256) {                   // scalar out: jj = (j1,j2)
        int j1 = n >> 4, j2 = n & 15;
        const float* u = U0_3 + (((j1 * 16 + j2) * 16 + i) * 23);
        const float* w = W0_3 + (size_t)e * 23 * NC + c;
        #pragma unroll
        for (int k = 0; k < 23; ++k) v += u[k] * w[k * NC];
      } else {                         // vector out: jj = (m,j1,j2)
        int vv = n - 256;
        int m = vv >> 8, j1 = (vv >> 4) & 15, j2 = vv & 15;
        const float* u = U1_3 + ((((m * 16 + j1) * 16 + j2) * 16 + i) * 23);
        const float* w = W1_3 + (size_t)e * 23 * NC + c;
        #pragma unroll
        for (int k = 0; k < 23; ++k) v += u[k] * w[k * NC];
      }
    }
  } else if (r < 170) {
    // rows (e): fused level-2 init (cols <1024) and level-1 init (cols >=1024)
    int e = r - 160;
    if (n < 256) {
      int j1 = n >> 4, j2 = n & 15;
      const float* u = U0_2 + (j1 * 16 + j2) * 4;
      const float* w = W0_2 + (size_t)e * 4 * NC + c;
      #pragma unroll
      for (int k = 0; k < 4; ++k) v += u[k] * w[k * NC];
    } else if (n < 1024) {
      int vv = n - 256;
      int m = vv >> 8, j1 = (vv >> 4) & 15, j2 = vv & 15;
      const float* u = U1_2 + ((m * 16 + j1) * 16 + j2) * 4;
      const float* w = W1_2 + (size_t)e * 4 * NC + c;
      #pragma unroll
      for (int k = 0; k < 4; ++k) v += u[k] * w[k * NC];
    } else if (n < 1040) {             // P1 scalar: j1 = n-1024, K0=1
      int j1 = n - 1024;
      v = U0_1[j1] * W0_1[(size_t)e * NC + c];
    } else {                           // P1 vector: (m,j1)
      int vv = n - 1040;
      int m = vv >> 4, j1 = vv & 15;
      v = U1_1[m * 16 + j1] * W1_1[(size_t)e * NC + c];
    }
  }
  size_t idx = ((((size_t)c * NT + t) * NKS + ks) * 32 + lane) * 16 + h;
  G[idx] = (__bf16)v;
}

// ---------------------------------------------------------------------------
// Kernel 2: per-(c, 32-row b-tile) WMMA GEMM + contraction epilogue.
// block = 128 threads (4 wave32). grid = (B/32, C).
// __launch_bounds__(128, 1): allow full VGPR allocation; A-fragments (96
// VGPRs) + 6 B-fragments (48) + 2 accumulators (16) stay register-resident.
// ---------------------------------------------------------------------------
__global__ void __launch_bounds__(128, 1) symcon_gemm(
    const float* __restrict__ A, const float* __restrict__ attrs,
    const __bf16* __restrict__ G, float* __restrict__ out)
{
  __shared__ __attribute__((aligned(32))) __bf16 Xf[2][NKS][32][16]; // A-frags
  __shared__ float Ash[32][16];
  __shared__ float accb[32][128]; // per b: [0:64]=Bt2(s,v) ; [64:128]=P1(s,v)

  const int b0 = blockIdx.x * 32;
  const int c = blockIdx.y;
  const int tid = threadIdx.x;

  // Stage 0: A tile into LDS.
  for (int idx = tid; idx < 512; idx += 128) {
    int b = idx >> 4, i = idx & 15;
    Ash[b][i] = A[((size_t)(b0 + b) * NC + c) * NI + i];
  }
  __syncthreads();

  // Stage 1: build X fragments (pre-swizzled WMMA A layout) in LDS.
  for (int idx = tid; idx < 2 * NKS * 32 * 16; idx += 128) {
    int h = idx & 15;
    int lane = (idx >> 4) & 31;
    int ks = (idx >> 9) % NKS;
    int sb = idx / (NKS * 32 * 16);
    int k = ks * 32 + klocal(lane, h);
    int m = lane & 15;
    int b = sb * 16 + m;
    float v = 0.f;
    if (k < 160) {
      int e = k >> 4, i = k & 15;
      v = attrs[(size_t)(b0 + b) * NE + e] * Ash[b][i];
    } else if (k < 170) {
      v = attrs[(size_t)(b0 + b) * NE + (k - 160)];
    }
    Xf[sb][ks][lane][h] = (__bf16)v;
  }
  __syncthreads();

  const int w = __builtin_amdgcn_readfirstlane(tid >> 5);  // uniform wave id
  const int lane = tid & 31;
  const int hi = lane >> 4;    // which M half this lane's acc rows are in
  const int nl = lane & 15;    // N column (== j2 for jj tiles)

  // Keep all 12 A-fragments register-resident across the whole n-tile loop.
  v16bf af[2][NKS];
  #pragma unroll
  for (int sb = 0; sb < 2; ++sb)
    #pragma unroll
    for (int ks = 0; ks < NKS; ++ks)
      af[sb][ks] = *reinterpret_cast<const v16bf*>(&Xf[sb][ks][lane][0]);

  // Per-lane A values for the j2 contraction: av[sb][r] = A[b_row, j2=nl].
  float av[2][8];
  #pragma unroll
  for (int sb = 0; sb < 2; ++sb)
    #pragma unroll
    for (int r = 0; r < 8; ++r)
      av[sb][r] = Ash[sb * 16 + r + 8 * hi][nl];

  for (int t = w; t < NT; t += 4) {
    const __bf16* gt = G + (((size_t)c * NT + t) * NKS * 32 + lane) * 16;

    // prefetch next tile's B-fragment block into caches
    if (t + 4 < NT)
      __builtin_prefetch(gt + (size_t)4 * NKS * 32 * 16, 0, 1);

    if (t < 64) {
      // Full-K tile: issue all 6 B-fragment loads (12x global_load_b128 in
      // one clause, single wait), then 12 back-to-back WMMAs.
      v16bf bf[NKS];
      #pragma unroll
      for (int ks = 0; ks < NKS; ++ks)
        bf[ks] = *reinterpret_cast<const v16bf*>(gt + (size_t)ks * 32 * 16);

      v8f acc0 = {};
      v8f acc1 = {};
      #pragma unroll
      for (int ks = 0; ks < NKS; ++ks) {
        acc0 = __builtin_amdgcn_wmma_f32_16x16x32_bf16(
            false, af[0][ks], false, bf[ks], (short)0, acc0, false, false);
        acc1 = __builtin_amdgcn_wmma_f32_16x16x32_bf16(
            false, af[1][ks], false, bf[ks], (short)0, acc1, false, false);
      }

      // jj tile: fixed (m,j1) -> slot t; contract over j2 (= lane column).
      #pragma unroll
      for (int sb = 0; sb < 2; ++sb) {
        v8f& acc = sb ? acc1 : acc0;
        #pragma unroll
        for (int r = 0; r < 8; ++r) {
          float p = acc[r] * av[sb][r];
          p += __shfl_xor(p, 1);
          p += __shfl_xor(p, 2);
          p += __shfl_xor(p, 4);
          p += __shfl_xor(p, 8);
          if (nl == 0) accb[sb * 16 + r + 8 * hi][t] = p;
        }
      }
    } else {
      // P1 tiles: only the attrs rows (k-step 5) are nonzero in G here.
      v16bf bf5 = *reinterpret_cast<const v16bf*>(gt + (size_t)5 * 32 * 16);
      v8f acc0 = {};
      v8f acc1 = {};
      acc0 = __builtin_amdgcn_wmma_f32_16x16x32_bf16(
          false, af[0][5], false, bf5, (short)0, acc0, false, false);
      acc1 = __builtin_amdgcn_wmma_f32_16x16x32_bf16(
          false, af[1][5], false, bf5, (short)0, acc1, false, false);

      // Value passes straight through; slot 64 + (t-64)*16 + j1.
      int base = 64 + (t - 64) * 16 + nl;
      #pragma unroll
      for (int sb = 0; sb < 2; ++sb) {
        v8f& acc = sb ? acc1 : acc0;
        #pragma unroll
        for (int r = 0; r < 8; ++r)
          accb[sb * 16 + r + 8 * hi][base] = acc[r];
      }
    }
  }
  __syncthreads();

  // Stage 3: out[b,c,s] = sum_j1 (Bt2[s*16+j1] + P1[64+s*16+j1]) * A[b,c,j1]
  int b = tid >> 2;
  int s = tid & 3;
  float sum = 0.f;
  #pragma unroll
  for (int j1 = 0; j1 < 16; ++j1)
    sum += (accb[b][s * 16 + j1] + accb[b][64 + s * 16 + j1]) * Ash[b][j1];
  out[((size_t)(b0 + b) * NC + c) * 4 + s] = sum;
}

// ---------------------------------------------------------------------------
extern "C" void kernel_launch(void* const* d_in, const int* in_sizes, int n_in,
                              void* d_out, int out_size, void* d_ws, size_t ws_size,
                              hipStream_t stream) {
  const float* A     = (const float*)d_in[0];
  const float* attrs = (const float*)d_in[1];
  const float* U0_1  = (const float*)d_in[2];
  const float* U0_2  = (const float*)d_in[3];
  const float* U0_3  = (const float*)d_in[4];
  const float* U1_1  = (const float*)d_in[5];
  const float* U1_2  = (const float*)d_in[6];
  const float* U1_3  = (const float*)d_in[7];
  const float* W0_1  = (const float*)d_in[8];
  const float* W0_2  = (const float*)d_in[9];
  const float* W0_3  = (const float*)d_in[10];
  const float* W1_1  = (const float*)d_in[11];
  const float* W1_2  = (const float*)d_in[12];
  const float* W1_3  = (const float*)d_in[13];

  const size_t g_elems = (size_t)NC * NT * NKS * 32 * 16;
  if (ws_size < g_elems * sizeof(__bf16)) return;  // workspace guard
  __bf16* G = (__bf16*)d_ws;

  symcon_g_precompute<<<dim3(NC * NT * NKS), dim3(512), 0, stream>>>(
      U0_1, U0_2, U0_3, U1_1, U1_2, U1_3,
      W0_1, W0_2, W0_3, W1_1, W1_2, W1_3, G);

  symcon_gemm<<<dim3(NB / 32, NC), dim3(128), 0, stream>>>(
      A, attrs, G, (float*)d_out);
}
